// QKVAttentionLegacy_7404523618828
// MI455X (gfx1250) — matrix operand order repaired and checked by reading, not compiled, not run on hardware
//
#include <hip/hip_runtime.h>
#include <math.h>

// ---------------------------------------------------------------------------
// Fused flash-attention for QKVAttentionLegacy on gfx1250 (CDNA5, wave32).
//   qkv:        (2, 3072, 2048) fp32   -> per head: Q/K/V rows of 64 x 2048
//   encoder_kv: (2, 2048,  512) fp32   -> per head: eK/eV rows of 64 x 512
//   out:        (2, 1024, 2048) fp32
// 32 heads, ch = 64, T = 2048, S = 512 (enc) + 2048 (self) = 2560.
// One block = 8 waves; each wave owns 16 query rows; fused online softmax.
// s-tile = 64. Row-sums via P x ones WMMA; row-max via packed-f16 shuffles.
// A-fragment reloads use contiguous 16B runs of the 16-bit A layout (b128).
// ---------------------------------------------------------------------------

typedef __attribute__((ext_vector_type(16))) _Float16     v16h;
typedef __attribute__((ext_vector_type(2)))  _Float16     v2h;
typedef __attribute__((ext_vector_type(8)))  float        v8f;
typedef __attribute__((ext_vector_type(4)))  unsigned int v4u;

union Frag16 {
    v16h v;
    _Float16 h[16];
    unsigned int w[8];
    v4u q[2];
};

union PkH {
    v2h h2;
    unsigned int u;
};

#define LOG2E 1.44269504088896340736f

__launch_bounds__(256)
__global__ void qkv_attn_flash_kernel(const float* __restrict__ qkv,
                                      const float* __restrict__ ekv,
                                      float* __restrict__ out)
{
    constexpr int T    = 2048;   // query length
    constexpr int SENC = 512;    // encoder kv length
    constexpr int STOT = 2560;   // total kv length
    constexpr int BN   = 64;     // s-tile

    __shared__ __align__(32) _Float16 lds_kT[BN][64];      // K tile, [s][c]
    __shared__ __align__(32) _Float16 lds_v [64][BN];      // V tile, [c][s]
    __shared__ __align__(32) _Float16 lds_p [8][16][BN];   // per-wave P tile

    const int tid  = threadIdx.x;
    const int lane = tid & 31;
    const int wave = tid >> 5;
    const int lrow = lane & 15;   // M row (A/C) or N col (B/C)
    const int half = lane >> 4;   // lane half selects K-range / row-group

    const int head = blockIdx.y;  // 0..31
    const int b2   = head >> 4;
    const int hib  = head & 15;
    const int t0   = blockIdx.x * 128 + wave * 16;

    const float* qbase  = qkv + (size_t)(b2 * 3072 + hib * 192) * T;
    const float* kbase  = qbase + (size_t)64  * T;
    const float* vbase  = qbase + (size_t)128 * T;
    const float* ekbase = ekv + (size_t)(b2 * 2048 + hib * 128) * SENC;
    const float* evbase = ekbase + (size_t)64 * SENC;

    // ---- Q A-fragments (16x32 f16, K = c), both scale factors folded (1/8).
    Frag16 aQ[2];
#pragma unroll
    for (int cb = 0; cb < 2; ++cb) {
#pragma unroll
        for (int i = 0; i < 16; ++i) {
            const int v = i >> 1, p = i & 1;
            const int k = half * 16 + i;
            const int c = cb * 32 + k;
            aQ[cb].h[i] = (_Float16)(qbase[(size_t)c * T + (t0 + lrow)] * 0.125f);
        }
    }

    // ---- all-ones B fragment for row-sum-by-matmul
    Frag16 bOnes;
#pragma unroll
    for (int i = 0; i < 16; ++i) bOnes.h[i] = (_Float16)1.0f;

    // ---- online-softmax state (rows replicated across the 16-lane half)
    v8f acc[4];
#pragma unroll
    for (int j = 0; j < 4; ++j)
#pragma unroll
        for (int r = 0; r < 8; ++r) acc[j][r] = 0.0f;

    float mrow[8], lsum[8];
#pragma unroll
    for (int r = 0; r < 8; ++r) { mrow[r] = -3.0e38f; lsum[r] = 0.0f; }

    // staging assignment: 4 threads per c-row, 16 consecutive s each
    const int sc = tid >> 2;
    const int so = (tid & 3) * 16;

    for (int sb = 0; sb < STOT; sb += BN) {
        __syncthreads();   // protect LDS tiles from previous iteration's readers

        // ---- stage K^T and V tiles (fp32 -> f16)
        {
            const bool enc = (sb < SENC);
            const float* kp = enc ? (ekbase + (size_t)sc * SENC + sb + so)
                                  : (kbase  + (size_t)sc * T + (sb - SENC) + so);
            const float* vp = enc ? (evbase + (size_t)sc * SENC + sb + so)
                                  : (vbase  + (size_t)sc * T + (sb - SENC) + so);
#pragma unroll
            for (int j = 0; j < 16; j += 2) {
                lds_kT[so + j][sc]     = (_Float16)kp[j];
                lds_kT[so + j + 1][sc] = (_Float16)kp[j + 1];
                v2h pv = { (_Float16)vp[j], (_Float16)vp[j + 1] };
                *(v2h*)&lds_v[sc][so + j] = pv;   // packed ds_store_b32
            }
            if (sb + BN < STOT) {   // prefetch next tile -> global_prefetch_b8
                const int sn = sb + BN;
                const bool e2 = (sn < SENC);
                const float* kp2 = e2 ? (ekbase + (size_t)sc * SENC + sn + so)
                                      : (kbase  + (size_t)sc * T + (sn - SENC) + so);
                const float* vp2 = e2 ? (evbase + (size_t)sc * SENC + sn + so)
                                      : (vbase  + (size_t)sc * T + (sn - SENC) + so);
                __builtin_prefetch(kp2, 0, 1);
                __builtin_prefetch(vp2, 0, 1);
            }
        }
        __syncthreads();

        // ---- S tile = Q * K^T  (four 16x16 n-tiles, K-dim = 64 = 2 chunks)
        v8f st[4];
#pragma unroll
        for (int nt = 0; nt < 4; ++nt) {
            const int sn = nt * 16 + lrow;          // this lane's s column
            Frag16 bK0, bK1;
            bK0.v = *(const v16h*)&lds_kT[sn][half * 16];       // c 0..31
            bK1.v = *(const v16h*)&lds_kT[sn][32 + half * 16];  // c 32..63
            v8f z = {0.f, 0.f, 0.f, 0.f, 0.f, 0.f, 0.f, 0.f};
            st[nt] = __builtin_amdgcn_wmma_f32_16x16x32_f16(
                false, aQ[0].v, false, bK0.v, (short)0, z, false, false);
            st[nt] = __builtin_amdgcn_wmma_f32_16x16x32_f16(
                false, aQ[1].v, false, bK1.v, (short)0, st[nt], false, false);
        }

        // ---- running max: local max per row, then packed-f16 xor-reduce
        //      (softmax is shift-invariant; f16-rounded max is a valid shift)
        PkH pk[4];
#pragma unroll
        for (int i = 0; i < 4; ++i) {
            const int r0 = 2 * i, r1 = 2 * i + 1;
            const float a = fmaxf(fmaxf(st[0][r0], st[1][r0]),
                                  fmaxf(st[2][r0], st[3][r0]));
            const float b = fmaxf(fmaxf(st[0][r1], st[1][r1]),
                                  fmaxf(st[2][r1], st[3][r1]));
            pk[i].h2 = (v2h){ (_Float16)a, (_Float16)b };
        }
#pragma unroll
        for (int mask = 1; mask <= 8; mask <<= 1) {
#pragma unroll
            for (int i = 0; i < 4; ++i) {
                PkH o;
                o.u = __shfl_xor(pk[i].u, mask, 32);
                pk[i].h2 = __builtin_elementwise_max(pk[i].h2, o.h2);
            }
        }

        float pr[4][8], alpha[8];
#pragma unroll
        for (int r = 0; r < 8; ++r) {
            const float mx = (float)pk[r >> 1].h2[r & 1];
            const float mnew = fmaxf(mrow[r], mx);
            alpha[r] = exp2f((mrow[r] - mnew) * LOG2E);
#pragma unroll
            for (int nt = 0; nt < 4; ++nt)
                pr[nt][r] = exp2f((st[nt][r] - mnew) * LOG2E);
            mrow[r] = mnew;
        }
#pragma unroll
        for (int j = 0; j < 4; ++j)
#pragma unroll
            for (int r = 0; r < 8; ++r) acc[j][r] *= alpha[r];

        // ---- C-layout -> A-layout for P via per-wave LDS bounce
#pragma unroll
        for (int r = 0; r < 8; ++r) {
            const int m = half * 8 + r;
#pragma unroll
            for (int nt = 0; nt < 4; ++nt)
                lds_p[wave][m][nt * 16 + lrow] = (_Float16)pr[nt][r];
        }
        asm volatile("s_wait_dscnt 0" ::: "memory");   // wave-internal LDS RAW

        // A layout is contiguous in 8-element (16B) runs: two b128 per chunk
        Frag16 aP[2];
#pragma unroll
        for (int kb = 0; kb < 2; ++kb) {
            aP[kb].q[0] = *(const v4u*)&lds_p[wave][lrow][kb * 32 + half * 16];
            aP[kb].q[1] = *(const v4u*)&lds_p[wave][lrow][kb * 32 + half * 16 + 8];
        }

        // ---- row sums as P x ones (replicated across lanes, no shuffles)
        {
            v8f z = {0.f, 0.f, 0.f, 0.f, 0.f, 0.f, 0.f, 0.f};
            v8f rs = __builtin_amdgcn_wmma_f32_16x16x32_f16(
                false, aP[0].v, false, bOnes.v, (short)0, z, false, false);
            rs = __builtin_amdgcn_wmma_f32_16x16x32_f16(
                false, aP[1].v, false, bOnes.v, (short)0, rs, false, false);
#pragma unroll
            for (int r = 0; r < 8; ++r)
                lsum[r] = lsum[r] * alpha[r] + rs[r];
        }

        // ---- O += P * V^T  (four 16x16 c-tiles, K-dim = 64 s = 2 chunks)
#pragma unroll
        for (int j = 0; j < 4; ++j) {
            Frag16 bV0, bV1;
            bV0.v = *(const v16h*)&lds_v[j * 16 + lrow][half * 16];       // s 0..31
            bV1.v = *(const v16h*)&lds_v[j * 16 + lrow][32 + half * 16];  // s 32..63
            acc[j] = __builtin_amdgcn_wmma_f32_16x16x32_f16(
                false, aP[0].v, false, bV0.v, (short)0, acc[j], false, false);
            acc[j] = __builtin_amdgcn_wmma_f32_16x16x32_f16(
                false, aP[1].v, false, bV1.v, (short)0, acc[j], false, false);
        }
    }

    // ---- normalize and store: out[b2][hib*64 + c][t]
    const size_t obase = (size_t)b2 * 1024 + (size_t)hib * 64;
#pragma unroll
    for (int j = 0; j < 4; ++j) {
        const int c = j * 16 + lrow;
        float* op = out + (obase + c) * (size_t)T + t0 + half * 8;
        float4 lo = make_float4(acc[j][0] / lsum[0], acc[j][1] / lsum[1],
                                acc[j][2] / lsum[2], acc[j][3] / lsum[3]);
        float4 hi = make_float4(acc[j][4] / lsum[4], acc[j][5] / lsum[5],
                                acc[j][6] / lsum[6], acc[j][7] / lsum[7]);
        *(float4*)(op)     = lo;
        *(float4*)(op + 4) = hi;
    }
}

extern "C" void kernel_launch(void* const* d_in, const int* in_sizes, int n_in,
                              void* d_out, int out_size, void* d_ws, size_t ws_size,
                              hipStream_t stream) {
    const float* qkv = (const float*)d_in[0];   // (2, 3072, 2048) fp32
    const float* ekv = (const float*)d_in[1];   // (2, 2048, 512) fp32
    float* out = (float*)d_out;                 // (2, 1024, 2048) fp32

    dim3 grid(2048 / 128, 32);   // 16 query tiles x 32 heads
    dim3 block(256);             // 8 waves; each wave = 16 query rows
    hipLaunchKernelGGL(qkv_attn_flash_kernel, grid, block, 0, stream,
                       qkv, ekv, out);
}
